// LocalAttention_24257975287925
// MI455X (gfx1250) — matrix operand order repaired and checked by reading, not compiled, not run on hardware
//
#include <hip/hip_runtime.h>
#include <hip/hip_bf16.h>

// CDNA5 / gfx1250 fused masked local attention with relative positions.
// One workgroup = 8 wave32 per (head, 256-block, 64-row query chunk).
// bf16 WMMA (v_wmma_f32_16x16x32_bf16) with f32 accumulate.
//
// Rel-position trick: rel[x,y] = y-x+257 depends only on the diagonal:
//  - rel-logits: GEMM Q @ Ek[2:770]^T over r' in [0,768), ds_add_f32
//    scatter-add along diagonals into S.
//  - rel-values: probabilities kept in BOTH unshifted (Wuns, for P@V) and
//    shifted (Wshift[x, r'=y+255-qi], for Wshift@EvT) form; all phase-B
//    fragment loads are aligned ds_load_b128.

typedef __bf16 bf16_t;
typedef __attribute__((ext_vector_type(16))) __bf16 v16bf;
typedef __attribute__((ext_vector_type(8)))  float  v8f;

#define BH_   16
#define L_    8192
#define D_    64
#define QS_   256     // query block
#define ML_   512     // memory window (flange 256 + block 256)
#define NB_   32      // L_/QS_
#define QC_   64      // query rows per workgroup
#define RW_   768     // shifted-rel width: r' = y+255-qi in [0,767]
#define NTHR_ 256

// LDS byte offsets (total 327680 B == 320 KB, the WGP maximum)
#define OFF_K    0        // bf16 [512][64]  = 65536   (phase A only)
#define OFF_Q    65536    // bf16 [64][64]   = 8192    (phase A only)
#define OFF_W    0        // bf16 [64][768]  = 98304   Wshift (overlays K+Q)
#define OFF_VT   98304    // bf16 [64][512]  = 65536   V transposed [d][y]
#define OFF_S    163840   // f32  [64][512]  = 131072  logits (dead after softmax)
#define OFF_EVT  163840   // bf16 [64][768]  = 98304   EvT (overlays S rows 0-47)
#define OFF_WU   262144   // bf16 [64][512]  = 65536   Wuns (rows 0-31 overlay S rows 48-63)
#define SMEM_BYTES 327680

union AB { v16bf v; bf16_t e[16]; uint4 q4[2]; };

__device__ inline v8f wmma_bf16(v16bf a, v16bf b, v8f c) {
  return __builtin_amdgcn_wmma_f32_16x16x32_bf16(
      /*neg_a*/false, a, /*neg_b*/false, b, /*c_mod*/(short)0, c,
      /*reuse_a*/false, /*reuse_b*/false);
}

// A fragment (16x32) from row-major bf16 matrix, pitch P (contiguous k order):
// lanes 0-15: M=row, K = k0+0..15; lanes 16-31: K = k0+16..31.
__device__ inline v16bf loadA(const bf16_t* M, int P, int row0, int k0, int lane) {
  int r  = row0 + (lane & 15);
  int kb = (lane >> 4) << 4;                 // 0 or 16
  const bf16_t* p = M + r * P + k0 + kb;
  AB u;
  u.q4[0] = *(const uint4*)(p);              // K = k0+kb   .. +7
  u.q4[1] = *(const uint4*)(p + 8);          // K = k0+kb+8 .. +15
  return u.v;
}

// B fragment (32x16) where value(k,n) = M[n*P + k]:
// lanes 0-15: N=n, K = k0+0..15 ; lanes 16-31: K = k0+16..31.
__device__ inline v16bf loadB_nt(const bf16_t* M, int P, int k0, int n0, int lane) {
  int nn = n0 + (lane & 15);
  int kb = (lane >> 4) << 4;                 // 0 or 16
  const bf16_t* p = M + nn * P + k0 + kb;
  AB u;
  u.q4[0] = *(const uint4*)(p);
  u.q4[1] = *(const uint4*)(p + 8);
  return u.v;
}

// Same layout, sourced from global fp32 (rel_k_emb rows contiguous in K=d).
__device__ inline v16bf loadB_nt_g(const float* G, int k0, int n0, int lane) {
  int nn = n0 + (lane & 15);
  int kb = (lane >> 4) << 4;
  const float* p = G + nn * 64 + k0 + kb;
  AB u;
#pragma unroll
  for (int e = 0; e < 16; ++e) u.e[e] = (bf16_t)p[e];
  return u.v;
}

// Masked softmax for a batch of 32 rows (8 lanes/row, reductions via shuffles).
// Writes normalized bf16 probabilities into Wuns (aligned).
__device__ inline void softmax_batch(float* sS, bf16_t* sWu, int rowbase,
                                     int tid, int qi0, int lo) {
  int row = rowbase + (tid >> 3);
  int y0  = (tid & 7) * 64;
  int qi  = qi0 + row;
  int hi  = qi + 256;                        // valid: lo <= y <= hi
  float* srow = sS + row * ML_;

  float m = -3.0e38f;
  for (int y = y0; y < y0 + 64; y += 4) {
    float4 f = *(const float4*)(srow + y);
    if (y + 0 >= lo && y + 0 <= hi) m = fmaxf(m, f.x);
    if (y + 1 >= lo && y + 1 <= hi) m = fmaxf(m, f.y);
    if (y + 2 >= lo && y + 2 <= hi) m = fmaxf(m, f.z);
    if (y + 3 >= lo && y + 3 <= hi) m = fmaxf(m, f.w);
  }
  m = fmaxf(m, __shfl_xor(m, 1));
  m = fmaxf(m, __shfl_xor(m, 2));
  m = fmaxf(m, __shfl_xor(m, 4));

  float s = 0.f;
  for (int y = y0; y < y0 + 64; y += 4) {
    float4 f = *(float4*)(srow + y);
    f.x = (y + 0 >= lo && y + 0 <= hi) ? __expf(f.x - m) : 0.f;
    f.y = (y + 1 >= lo && y + 1 <= hi) ? __expf(f.y - m) : 0.f;
    f.z = (y + 2 >= lo && y + 2 <= hi) ? __expf(f.z - m) : 0.f;
    f.w = (y + 3 >= lo && y + 3 <= hi) ? __expf(f.w - m) : 0.f;
    *(float4*)(srow + y) = f;
    s += f.x + f.y + f.z + f.w;
  }
  s += __shfl_xor(s, 1);
  s += __shfl_xor(s, 2);
  s += __shfl_xor(s, 4);
  float inv = 1.0f / s;

  bf16_t* wrow = sWu + row * ML_;
  for (int y = y0; y < y0 + 64; y += 4) {
    float4 f = *(const float4*)(srow + y);
    wrow[y + 0] = (bf16_t)(f.x * inv);
    wrow[y + 1] = (bf16_t)(f.y * inv);
    wrow[y + 2] = (bf16_t)(f.z * inv);
    wrow[y + 3] = (bf16_t)(f.w * inv);
  }
}

__global__ __launch_bounds__(NTHR_, 1)
void local_attn_kernel(const float* __restrict__ q, const float* __restrict__ k,
                       const float* __restrict__ v, const float* __restrict__ ek,
                       const float* __restrict__ ev, float* __restrict__ out) {
  extern __shared__ char smem[];
  bf16_t* sK   = (bf16_t*)(smem + OFF_K);
  bf16_t* sQ   = (bf16_t*)(smem + OFF_Q);
  bf16_t* sW   = (bf16_t*)(smem + OFF_W);    // Wshift
  bf16_t* sVT  = (bf16_t*)(smem + OFF_VT);
  float*  sS   = (float*)(smem + OFF_S);
  bf16_t* sEvT = (bf16_t*)(smem + OFF_EVT);
  bf16_t* sWu  = (bf16_t*)(smem + OFF_WU);   // Wuns

  const int tid  = threadIdx.x;
  const int lane = tid & 31;
  const int wave = tid >> 5;

  const int blk = blockIdx.x;            // ((bh*32 + n)*4 + c)
  const int c   = blk & 3;
  const int n   = (blk >> 2) & 31;
  const int bh  = blk >> 7;

  const int  qi0 = c * QC_;              // base query index within the 256-block
  const long ql0 = (long)n * QS_ + qi0;  // absolute query row of chunk row 0
  const long kl0 = (long)n * QS_ - 256;  // absolute key row of memory index 0
  const int  lo  = (n == 0) ? 256 : 0;   // first valid memory index

  // ---------- Load Q (scaled by D^-0.5) / K / V(transposed) as bf16 ----------
  {
    const float* qs = q + ((long)bh * L_ + ql0) * D_;
    for (int i = tid * 4; i < QC_ * D_; i += NTHR_ * 4) {
      float4 f = *(const float4*)(qs + i);
      sQ[i + 0] = (bf16_t)(f.x * 0.125f);
      sQ[i + 1] = (bf16_t)(f.y * 0.125f);
      sQ[i + 2] = (bf16_t)(f.z * 0.125f);
      sQ[i + 3] = (bf16_t)(f.w * 0.125f);
    }
    for (int i = tid * 4; i < ML_ * D_; i += NTHR_ * 4) {
      int  row = i >> 6;                 // memory index mi
      int  col = i & 63;
      long lk  = kl0 + row;
      float4 fk = {0.f, 0.f, 0.f, 0.f};
      float4 fv = {0.f, 0.f, 0.f, 0.f};
      if (lk >= 0) {                     // left pad region of block 0 is zeros
        long base = ((long)bh * L_ + lk) * D_ + col;
        fk = *(const float4*)(k + base);
        fv = *(const float4*)(v + base);
      }
      sK[i + 0] = (bf16_t)fk.x; sK[i + 1] = (bf16_t)fk.y;
      sK[i + 2] = (bf16_t)fk.z; sK[i + 3] = (bf16_t)fk.w;
      sVT[(col + 0) * ML_ + row] = (bf16_t)fv.x;   // V stored [d][y]
      sVT[(col + 1) * ML_ + row] = (bf16_t)fv.y;
      sVT[(col + 2) * ML_ + row] = (bf16_t)fv.z;
      sVT[(col + 3) * ML_ + row] = (bf16_t)fv.w;
    }
  }
  __syncthreads();

  // ---------- Preload all 8 Q A-fragments (reused by QK^T and QE) ----------
  v16bf aq[4][2];
#pragma unroll
  for (int r = 0; r < 4; ++r)
#pragma unroll
    for (int kk = 0; kk < 2; ++kk)
      aq[r][kk] = loadA(sQ, D_, r << 4, kk << 5, lane);

  // ---------- Phase A1: S = Q @ K^T (64x512), B shared over 4 row tiles ----
#pragma unroll
  for (int j = 0; j < 4; ++j) {
    int col0 = (wave << 4) + (j << 7);
    v16bf b0 = loadB_nt(sK, D_, 0,  col0, lane);
    v16bf b1 = loadB_nt(sK, D_, 32, col0, lane);
#pragma unroll
    for (int r = 0; r < 4; ++r) {
      v8f acc = {0.f, 0.f, 0.f, 0.f, 0.f, 0.f, 0.f, 0.f};
      acc = wmma_bf16(aq[r][0], b0, acc);
      acc = wmma_bf16(aq[r][1], b1, acc);
      int y  = col0 + (lane & 15);
      int xr = (r << 4) + ((lane >> 4) << 3);
#pragma unroll
      for (int u = 0; u < 8; ++u) sS[(xr + u) * ML_ + y] = acc[u];
    }
  }
  __syncthreads();

  // ---------- Phase A2: QE = Q @ Ek[2:770]^T, ds_add_f32 on diagonals -----
  // S[x, r' + qi - 255] += QE[x, r']
#pragma unroll
  for (int j = 0; j < 6; ++j) {
    int r0 = (wave + (j << 3)) << 4;     // 48 r'-tiles over [0,768)
    v16bf b0 = loadB_nt_g(ek + 2 * 64, 0,  r0, lane);
    v16bf b1 = loadB_nt_g(ek + 2 * 64, 32, r0, lane);
#pragma unroll
    for (int tr = 0; tr < 4; ++tr) {
      v8f acc = {0.f, 0.f, 0.f, 0.f, 0.f, 0.f, 0.f, 0.f};
      acc = wmma_bf16(aq[tr][0], b0, acc);
      acc = wmma_bf16(aq[tr][1], b1, acc);
      int rr = r0 + (lane & 15);
      int xr = (tr << 4) + ((lane >> 4) << 3);
#pragma unroll
      for (int u = 0; u < 8; ++u) {
        int x = xr + u;
        int y = rr + (qi0 + x) - 255;
        if ((unsigned)y < (unsigned)ML_) atomicAdd(&sS[x * ML_ + y], acc[u]);
      }
    }
  }
  __syncthreads();

  // ---------- Masked softmax -> Wuns (two batches; Wuns rows 0-31 overlay
  // S rows 48-63, so consume rows 32-63 first) ----------
  softmax_batch(sS, sWu, 32, tid, qi0, lo);
  __syncthreads();
  softmax_batch(sS, sWu, 0, tid, qi0, lo);
  __syncthreads();   // sS / sK / sQ all dead from here

  // ---------- Build Wshift from Wuns (also zero-fills outside the band) ----
  {
    int x  = tid >> 2;                   // one row per 4 threads
    int p0 = (tid & 3) * 192;
    int qi = qi0 + x;
    const bf16_t* wu = sWu + x * ML_;
    bf16_t* ws = sW + x * RW_;
    for (int r = p0; r < p0 + 192; ++r) {
      int y = r + qi - 255;
      ws[r] = ((unsigned)y < (unsigned)ML_) ? wu[y] : (bf16_t)0.f;
    }
  }
  // ---------- Stage EvT: sEvT[d][r'] = (bf16) ev[r'+2][d] ----------
  for (int i = tid * 4; i < RW_ * D_; i += NTHR_ * 4) {
    int r = i >> 6;
    int d = i & 63;
    float4 f = *(const float4*)(ev + (long)(r + 2) * 64 + d);
    sEvT[(d + 0) * RW_ + r] = (bf16_t)f.x;
    sEvT[(d + 1) * RW_ + r] = (bf16_t)f.y;
    sEvT[(d + 2) * RW_ + r] = (bf16_t)f.z;
    sEvT[(d + 3) * RW_ + r] = (bf16_t)f.w;
  }
  __syncthreads();

  // ---------- Phase B: out = Wuns@V + Wshift@EvT (all b128 fragments) ------
  // Each wave: fixed d0, two row tiles (rbase, rbase+32); B shared across both.
  float* outp  = out + ((long)bh * L_ + ql0) * D_;
  const int d0    = (wave & 3) << 4;
  const int rbase = (wave >> 2) << 4;
  v8f acc0 = {0.f, 0.f, 0.f, 0.f, 0.f, 0.f, 0.f, 0.f};
  v8f acc1 = {0.f, 0.f, 0.f, 0.f, 0.f, 0.f, 0.f, 0.f};
#pragma unroll
  for (int k0 = 0; k0 < ML_; k0 += 32) {          // P @ V
    v16bf b = loadB_nt(sVT, ML_, k0, d0, lane);
    acc0 = wmma_bf16(loadA(sWu, ML_, rbase,      k0, lane), b, acc0);
    acc1 = wmma_bf16(loadA(sWu, ML_, rbase + 32, k0, lane), b, acc1);
  }
#pragma unroll
  for (int k0 = 0; k0 < RW_; k0 += 32) {          // Wshift @ EvT
    v16bf b = loadB_nt(sEvT, RW_, k0, d0, lane);
    acc0 = wmma_bf16(loadA(sW, RW_, rbase,      k0, lane), b, acc0);
    acc1 = wmma_bf16(loadA(sW, RW_, rbase + 32, k0, lane), b, acc1);
  }
  {
    int dc = d0 + (lane & 15);
    int xr = rbase + ((lane >> 4) << 3);
#pragma unroll
    for (int u = 0; u < 8; ++u) {
      outp[(long)(xr + u) * D_ + dc]      = acc0[u];
      outp[(long)(xr + u + 32) * D_ + dc] = acc1[u];
    }
  }
}

extern "C" void kernel_launch(void* const* d_in, const int* in_sizes, int n_in,
                              void* d_out, int out_size, void* d_ws, size_t ws_size,
                              hipStream_t stream) {
  (void)in_sizes; (void)n_in; (void)out_size; (void)d_ws; (void)ws_size;
  const float* q  = (const float*)d_in[0];
  const float* k  = (const float*)d_in[1];
  const float* v  = (const float*)d_in[2];
  const float* ek = (const float*)d_in[3];  // rel_k_emb (1027, 64)
  const float* ev = (const float*)d_in[4];  // rel_v_emb (1027, 64)

  hipFuncSetAttribute((const void*)local_attn_kernel,
                      hipFuncAttributeMaxDynamicSharedMemorySize, SMEM_BYTES);

  dim3 grid(BH_ * NB_ * (QS_ / QC_));     // 16 * 32 * 4 = 2048 workgroups
  dim3 block(NTHR_);                      // 8 wave32
  local_attn_kernel<<<grid, block, SMEM_BYTES, stream>>>(q, k, v, ek, ev,
                                                         (float*)d_out);
}
